// LayerNormGRUCell_4776003633435
// MI455X (gfx1250) — compile-verified, matrix operand order checked by reading, not run on hardware
//
#include <hip/hip_runtime.h>

typedef __attribute__((ext_vector_type(16))) __bf16          v16bf;
typedef __attribute__((ext_vector_type(8)))  float           v8f;
typedef __attribute__((ext_vector_type(4)))  float           f32x4;
typedef __attribute__((ext_vector_type(4)))  unsigned short  u16x4;

#define B_TOTAL 16384
#define K_DIM   512
#define ROWS    32
#define THREADS 256
#define LN_EPS  1e-5f

// LDS carve (bytes):
#define OFF_SX    0        // 32x512 bf16 fragment-major   = 32768
#define OFF_SH    32768    // 32x512 bf16 fragment-major   = 32768
#define OFF_I2H   65536    // 32 rows x (1024+8) bf16      = 66048, stride 2064
#define OFF_H2H   131584   // 66048
#define OFF_HATA  197632   // 32 rows x (512+8) bf16       = 33280, stride 1040
#define OFF_HATB  230912   // 33280
#define OFF_STATS 264192   // 32 rows x 8 f32              = 1024
#define SMEM_BYTES 265216

// d_ws layout (bytes): bf16 fragment-major weights
//   W_i2h @0 (1 MiB), W_h2h @1 MiB, W_hatW @2 MiB (512 KiB), W_hatU @2.5 MiB. Total 3 MiB.
#define WOFF_I2H  0u
#define WOFF_H2H  1048576u
#define WOFF_HATW 2097152u
#define WOFF_HATU 2621440u

__device__ __forceinline__ unsigned short f2bf(float f) {
  unsigned u = __builtin_bit_cast(unsigned, f);
  u += 0x7FFFu + ((u >> 16) & 1u);          // round-to-nearest-even
  return (unsigned short)(u >> 16);
}
__device__ __forceinline__ float bf2f(unsigned short s) {
  unsigned u = ((unsigned)s) << 16;
  return __builtin_bit_cast(float, u);
}
__device__ __forceinline__ float sigm(float x) { return 1.0f / (1.0f + __expf(-x)); }
__device__ __forceinline__ float tanh_fast(float x) { return 1.0f - 2.0f / (1.0f + __expf(2.0f * x)); }

// ---------------------------------------------------------------------------
// Kernel 0: repack f32 weights -> bf16 fragment-major blocks in d_ws.
// Fragment block for (nTile, kTile) is 1024 B; lane L holds W[n0+(L&15)][k0+16*(L>>4) .. +15].
// ---------------------------------------------------------------------------
__global__ __launch_bounds__(THREADS) void repack_weights(
    const float* __restrict__ Wi, const float* __restrict__ Wh,
    const float* __restrict__ Ww, const float* __restrict__ Wu,
    unsigned short* __restrict__ wB)
{
  int o4 = blockIdx.x * THREADS + threadIdx.x;   // quad index, 0..393215
  const float* src;
  unsigned short* dst;
  int rel;
  if (o4 < 131072)      { src = Wi; dst = wB;            rel = o4; }
  else if (o4 < 262144) { src = Wh; dst = wB + 524288;   rel = o4 - 131072; }
  else if (o4 < 327680) { src = Ww; dst = wB + 1048576;  rel = o4 - 262144; }
  else                  { src = Wu; dst = wB + 1310720;  rel = o4 - 327680; }

  int o    = rel << 2;
  int tile = o >> 9;
  int j    = o & 511;
  int L    = j >> 4;
  int e    = j & 15;
  int tn   = tile >> 4;          // N-tile
  int kt   = tile & 15;          // K-tile
  int n    = (tn << 4) + (L & 15);
  int k    = (kt << 5) + ((L >> 4) << 4) + e;

  f32x4 f = *(const f32x4*)(src + (size_t)n * K_DIM + k);
  u16x4 p;
  p.x = f2bf(f.x); p.y = f2bf(f.y); p.z = f2bf(f.z); p.w = f2bf(f.w);
  *(u16x4*)(dst + o) = p;
}

// ---------------------------------------------------------------------------
// Kernel 1: fused GEMMx4 (bf16 WMMA) + dual LayerNorm + GRU gate epilogue.
// One workgroup = 32 rows of the batch; everything stays in LDS.
// ---------------------------------------------------------------------------
__global__ __launch_bounds__(THREADS) void fused_gru(
    const float* __restrict__ x,  const float* __restrict__ h,
    const float* __restrict__ b_i2h,  const float* __restrict__ b_h2h,
    const float* __restrict__ b_hatW, const float* __restrict__ b_hatU,
    const unsigned short* __restrict__ wB, float* __restrict__ out)
{
  extern __shared__ char smem[];
  char*  sX    = smem + OFF_SX;
  char*  sH    = smem + OFF_SH;
  char*  sI2H  = smem + OFF_I2H;
  char*  sH2H  = smem + OFF_H2H;
  char*  sHatA = smem + OFF_HATA;
  char*  sHatB = smem + OFF_HATB;
  float* stats = (float*)(smem + OFF_STATS);

  const int tid  = threadIdx.x;
  const int lane = tid & 31;
  const int wave = tid >> 5;
  const int row0 = blockIdx.x * ROWS;

  // zero per-row stats (32 rows x 8 accumulators)
  stats[tid] = 0.0f;

  // ---- Stage x/h tiles into LDS as bf16, fragment-major (same layout as B) ----
  for (int q = tid; q < (ROWS * K_DIM / 4); q += THREADS) {
    int o    = q << 2;
    int tile = o >> 9;
    int j    = o & 511;
    int L    = j >> 4;
    int e    = j & 15;
    int mt   = tile >> 4;
    int kt   = tile & 15;
    int row  = (mt << 4) + (L & 15);
    int col  = (kt << 5) + ((L >> 4) << 4) + e;
    size_t gidx = (size_t)(row0 + row) * K_DIM + col;
    f32x4 vx = *(const f32x4*)(x + gidx);
    f32x4 vh = *(const f32x4*)(h + gidx);
    u16x4 px, ph;
    px.x = f2bf(vx.x); px.y = f2bf(vx.y); px.z = f2bf(vx.z); px.w = f2bf(vx.w);
    ph.x = f2bf(vh.x); ph.y = f2bf(vh.y); ph.z = f2bf(vh.z); ph.w = f2bf(vh.w);
    *(u16x4*)(sX + (o << 1)) = px;
    *(u16x4*)(sH + (o << 1)) = ph;
  }
  __syncthreads();

  // ---- GEMM phase: 48 jobs, each = 2 M-tiles x 4 N-tiles, K=512 ----
  // jobs 0-15: i2h (x @ W_i2h^T), 16-31: h2h, 32-39: hatA, 40-47: hatB
  for (int g = wave; g < 48; g += 8) {
    const char* aBase;
    const char* wBase;
    const float* bias;
    char* oBuf;
    int oStride, nt0;
    if (g < 16)      { aBase = sX; wBase = (const char*)wB + WOFF_I2H;  bias = b_i2h;  oBuf = sI2H;  oStride = 2064; nt0 = g << 2; }
    else if (g < 32) { aBase = sH; wBase = (const char*)wB + WOFF_H2H;  bias = b_h2h;  oBuf = sH2H;  oStride = 2064; nt0 = (g - 16) << 2; }
    else if (g < 40) { aBase = sX; wBase = (const char*)wB + WOFF_HATW; bias = b_hatW; oBuf = sHatA; oStride = 1040; nt0 = (g - 32) << 2; }
    else             { aBase = sH; wBase = (const char*)wB + WOFF_HATU; bias = b_hatU; oBuf = sHatB; oStride = 1040; nt0 = (g - 40) << 2; }

    const char* aL = aBase + (lane << 5);                              // A frags, mTile 0/1 at +0 / +16KB
    const char* wL = wBase + ((size_t)(nt0 << 4) << 10) + (lane << 5); // B frags for 4 consecutive N-tiles

    v8f acc[2][4] = {};

    // Software-pipelined K loop: prefetch kt+1 fragments while WMMAs consume kt.
    v16bf a0c = *(const v16bf*)(aL);
    v16bf a1c = *(const v16bf*)(aL + (16 << 10));
    v16bf b0c = *(const v16bf*)(wL);
    v16bf b1c = *(const v16bf*)(wL + (16 << 10));
    v16bf b2c = *(const v16bf*)(wL + (32 << 10));
    v16bf b3c = *(const v16bf*)(wL + (48 << 10));

    #pragma unroll 4
    for (int kt = 0; kt < 16; ++kt) {
      int kn = (kt + 1) & 15;     // wrap: final prefetch is dead and eliminated
      v16bf a0n = *(const v16bf*)(aL + (kn << 10));
      v16bf a1n = *(const v16bf*)(aL + ((16 + kn) << 10));
      v16bf b0n = *(const v16bf*)(wL + (kn << 10));
      v16bf b1n = *(const v16bf*)(wL + ((16 + kn) << 10));
      v16bf b2n = *(const v16bf*)(wL + ((32 + kn) << 10));
      v16bf b3n = *(const v16bf*)(wL + ((48 + kn) << 10));

      acc[0][0] = __builtin_amdgcn_wmma_f32_16x16x32_bf16(false, a0c, false, b0c, (short)0, acc[0][0], false, false);
      acc[1][0] = __builtin_amdgcn_wmma_f32_16x16x32_bf16(false, a1c, false, b0c, (short)0, acc[1][0], false, false);
      acc[0][1] = __builtin_amdgcn_wmma_f32_16x16x32_bf16(false, a0c, false, b1c, (short)0, acc[0][1], false, false);
      acc[1][1] = __builtin_amdgcn_wmma_f32_16x16x32_bf16(false, a1c, false, b1c, (short)0, acc[1][1], false, false);
      acc[0][2] = __builtin_amdgcn_wmma_f32_16x16x32_bf16(false, a0c, false, b2c, (short)0, acc[0][2], false, false);
      acc[1][2] = __builtin_amdgcn_wmma_f32_16x16x32_bf16(false, a1c, false, b2c, (short)0, acc[1][2], false, false);
      acc[0][3] = __builtin_amdgcn_wmma_f32_16x16x32_bf16(false, a0c, false, b3c, (short)0, acc[0][3], false, false);
      acc[1][3] = __builtin_amdgcn_wmma_f32_16x16x32_bf16(false, a1c, false, b3c, (short)0, acc[1][3], false, false);

      a0c = a0n; a1c = a1n;
      b0c = b0n; b1c = b1n; b2c = b2n; b3c = b3n;
    }

    // bias add + bf16 store to LDS (C layout: VGPR v -> row v + 8*(lane>>4), col = lane&15)
    #pragma unroll
    for (int j = 0; j < 4; ++j) {
      int col = ((nt0 + j) << 4) + (lane & 15);
      float bv = bias[col];
      #pragma unroll
      for (int m = 0; m < 2; ++m) {
        int rbase = (m << 4) + ((lane >> 4) << 3);
        #pragma unroll
        for (int v = 0; v < 8; ++v) {
          float f = acc[m][j][v] + bv;
          *(unsigned short*)(oBuf + (rbase + v) * oStride + (col << 1)) = f2bf(f);
        }
      }
    }
  }
  __syncthreads();

  // ---- Row statistics: wave w covers segment w of every row (conflict-free) ----
  {
    int row = tid & 31;
    int seg = tid >> 5;            // 0..7
    const char* r1p = sI2H  + row * 2064;
    const char* r2p = sH2H  + row * 2064;
    const char* r3p = sHatA + row * 1040;
    const char* r4p = sHatB + row * 1040;
    float s0 = 0, q0 = 0, s1 = 0, q1 = 0, s2 = 0, q2 = 0, s3 = 0, q3 = 0;
    for (int i = 0; i < 32; ++i) {          // 128 cols of i2h/h2h per thread
      u16x4 a = *(const u16x4*)(r1p + (((seg << 7) + (i << 2)) << 1));
      u16x4 b = *(const u16x4*)(r2p + (((seg << 7) + (i << 2)) << 1));
      float f;
      f = bf2f(a.x); s0 += f; q0 += f * f;  f = bf2f(a.y); s0 += f; q0 += f * f;
      f = bf2f(a.z); s0 += f; q0 += f * f;  f = bf2f(a.w); s0 += f; q0 += f * f;
      f = bf2f(b.x); s1 += f; q1 += f * f;  f = bf2f(b.y); s1 += f; q1 += f * f;
      f = bf2f(b.z); s1 += f; q1 += f * f;  f = bf2f(b.w); s1 += f; q1 += f * f;
    }
    for (int i = 0; i < 16; ++i) {          // 64 cols of hatA/hatB per thread
      u16x4 a = *(const u16x4*)(r3p + (((seg << 6) + (i << 2)) << 1));
      u16x4 b = *(const u16x4*)(r4p + (((seg << 6) + (i << 2)) << 1));
      float f;
      f = bf2f(a.x); s2 += f; q2 += f * f;  f = bf2f(a.y); s2 += f; q2 += f * f;
      f = bf2f(a.z); s2 += f; q2 += f * f;  f = bf2f(a.w); s2 += f; q2 += f * f;
      f = bf2f(b.x); s3 += f; q3 += f * f;  f = bf2f(b.y); s3 += f; q3 += f * f;
      f = bf2f(b.z); s3 += f; q3 += f * f;  f = bf2f(b.w); s3 += f; q3 += f * f;
    }
    float* st = stats + (row << 3);
    atomicAdd(st + 0, s0); atomicAdd(st + 1, q0);
    atomicAdd(st + 2, s1); atomicAdd(st + 3, q1);
    atomicAdd(st + 4, s2); atomicAdd(st + 5, q2);
    atomicAdd(st + 6, s3); atomicAdd(st + 7, q3);
  }
  __syncthreads();

  // ---- Epilogue: LN + sigmoid gates + tanh candidate + blend ----
  for (int idx = tid; idx < ROWS * 512; idx += THREADS) {
    int row = idx >> 9;
    int n   = idx & 511;
    const float* st = stats + (row << 3);
    float m1 = st[0] * (1.0f / 1024.0f);
    float r1 = rsqrtf(st[1] * (1.0f / 1024.0f) - m1 * m1 + LN_EPS);
    float m2 = st[2] * (1.0f / 1024.0f);
    float r2 = rsqrtf(st[3] * (1.0f / 1024.0f) - m2 * m2 + LN_EPS);
    float ma = st[4] * (1.0f / 512.0f);
    float ra = rsqrtf(st[5] * (1.0f / 512.0f) - ma * ma + LN_EPS);
    float mb = st[6] * (1.0f / 512.0f);
    float rb = rsqrtf(st[7] * (1.0f / 512.0f) - mb * mb + LN_EPS);

    const char* ri = sI2H + row * 2064;
    const char* rh = sH2H + row * 2064;
    float iz = (bf2f(*(const unsigned short*)(ri + (n << 1)))          - m1) * r1;
    float ir = (bf2f(*(const unsigned short*)(ri + ((n + 512) << 1)))  - m1) * r1;
    float hz = (bf2f(*(const unsigned short*)(rh + (n << 1)))          - m2) * r2;
    float hr = (bf2f(*(const unsigned short*)(rh + ((n + 512) << 1)))  - m2) * r2;
    float z  = sigm(iz + hz);
    float r  = sigm(ir + hr);
    float ha = (bf2f(*(const unsigned short*)(sHatA + row * 1040 + (n << 1))) - ma) * ra;
    float hb = (bf2f(*(const unsigned short*)(sHatB + row * 1040 + (n << 1))) - mb) * rb;
    float hh = tanh_fast(ha + r * hb);

    size_t gi = (size_t)(row0 + row) * 512 + n;
    float hp  = h[gi];
    out[gi] = (1.0f - z) * hp + z * hh;
  }
}

extern "C" void kernel_launch(void* const* d_in, const int* in_sizes, int n_in,
                              void* d_out, int out_size, void* d_ws, size_t ws_size,
                              hipStream_t stream) {
  (void)in_sizes; (void)n_in; (void)out_size; (void)ws_size;
  const float* x     = (const float*)d_in[0];
  const float* h     = (const float*)d_in[1];
  const float* Wi2h  = (const float*)d_in[2];
  const float* bi2h  = (const float*)d_in[3];
  const float* Wh2h  = (const float*)d_in[4];
  const float* bh2h  = (const float*)d_in[5];
  const float* WhatW = (const float*)d_in[6];
  const float* bhatW = (const float*)d_in[7];
  const float* WhatU = (const float*)d_in[8];
  const float* bhatU = (const float*)d_in[9];
  unsigned short* wB = (unsigned short*)d_ws;   // needs 3 MiB of workspace

  repack_weights<<<1536, THREADS, 0, stream>>>(Wi2h, Wh2h, WhatW, WhatU, wB);
  fused_gru<<<B_TOTAL / ROWS, THREADS, SMEM_BYTES, stream>>>(
      x, h, bi2h, bh2h, bhatW, bhatU, wB, (float*)d_out);
}
